// Classifier_oneGCN_GAP_43765716746306
// MI455X (gfx1250) — matrix operand order repaired and checked by reading, not compiled, not run on hardware
//
#include <hip/hip_runtime.h>
#include <hip/hip_bf16.h>

// ---------------- problem constants (match reference) ----------------
#define NODES  20000
#define EDGES  320000
#define GRAPHS 64
#define HD     256          // IN == H == 256
#define NCLS   2
#define NPAD   20096        // 157 * 128 (row padding for WMMA GEMM1)
#define GEMM1_BLOCKS 157

typedef float  v8f    __attribute__((ext_vector_type(8)));
typedef __bf16 v16bf  __attribute__((ext_vector_type(16)));
typedef __bf16 bf16x8 __attribute__((ext_vector_type(8)));

union AFrag {
    struct { bf16x8 lo; bf16x8 hi; } h;
    v16bf v;
};

// ---------------- utility kernels ----------------
__global__ void zero_f32(float* __restrict__ p, int n) {
    int i = blockIdx.x * blockDim.x + threadIdx.x;
    if (i < n) p[i] = 0.0f;
}

__global__ void degree_kernel(const int* __restrict__ src, const int* __restrict__ dst,
                              float* __restrict__ deg_out, float* __restrict__ deg_in, int nE) {
    int e = blockIdx.x * blockDim.x + threadIdx.x;
    if (e >= nE) return;
    atomicAdd(&deg_out[src[e]], 1.0f);
    atomicAdd(&deg_in[dst[e]], 1.0f);
}

// h * rsqrt(max(deg_out,1)) -> bf16, padded rows zeroed
__global__ void convert_h_kernel(const float* __restrict__ h, const float* __restrict__ deg_out,
                                 __bf16* __restrict__ hbf) {
    int row = blockIdx.x;            // 0..NPAD-1
    int col = threadIdx.x;           // 0..255
    float v = 0.0f;
    if (row < NODES) {
        float s = rsqrtf(fmaxf(deg_out[row], 1.0f));
        v = h[(size_t)row * HD + col] * s;
    }
    hbf[(size_t)row * HD + col] = (__bf16)v;
}

// W_conv (KxN) -> Wct (NxK bf16), W1 (KxN) -> W1t (NxK bf16)
__global__ void convert_w_kernel(const float* __restrict__ Wc, const float* __restrict__ W1,
                                 __bf16* __restrict__ Wct, __bf16* __restrict__ W1t) {
    int i = blockIdx.x * blockDim.x + threadIdx.x;   // 0 .. 2*65536-1
    int which = i >> 16;
    int j = i & 0xFFFF;
    int k = j >> 8, n = j & 255;
    if (which == 0) Wct[(size_t)n * HD + k] = (__bf16)Wc[(size_t)k * HD + n];
    else            W1t[(size_t)n * HD + k] = (__bf16)W1[(size_t)k * HD + n];
}

// ---------------- WMMA GEMM1: hn = hbf @ Wct^T   (NPAD x 256) ----------------
// block = 512 threads = 16 waves; tile = 128 rows x 256 cols
// wave (wy,wx): wy in 0..7 -> 16-row slice, wx in 0..1 -> 128-col half
__global__ __launch_bounds__(512)
void gemm_conv_wmma(const __bf16* __restrict__ A, const __bf16* __restrict__ Bt,
                    float* __restrict__ Cout, int nRows) {
    const int tid  = threadIdx.x;
    const int lane = tid & 31;
    const int w    = tid >> 5;
    const int wy   = w & 7;
    const int wx   = w >> 3;
    const int r0   = blockIdx.x * 128 + wy * 16;
    const int c0   = wx * 128;
    const int lh   = lane >> 4;    // 0/1: which K half this lane holds
    const int lm   = lane & 15;

    v8f acc[8] = {};

    const __bf16* aRow = A + (size_t)(r0 + lm) * HD;
    #pragma unroll
    for (int k0 = 0; k0 < HD; k0 += 32) {
        if (k0 + 64 < HD) __builtin_prefetch(aRow + k0 + 64, 0, 1);
        // A fragment: lane holds 16 contiguous K (k = 16*lh + s), matching B
        AFrag af;
        af.h.lo = *(const bf16x8*)(aRow + k0 + lh * 16);
        af.h.hi = *(const bf16x8*)(aRow + k0 + lh * 16 + 8);
        v16bf a = af.v;
        #pragma unroll
        for (int nt = 0; nt < 8; ++nt) {
            // B fragment: lane holds 16 contiguous K of column (c0+nt*16+lm)
            const __bf16* bp = Bt + (size_t)(c0 + nt * 16 + lm) * HD + k0 + lh * 16;
            v16bf b = *(const v16bf*)bp;
            acc[nt] = __builtin_amdgcn_wmma_f32_16x16x32_bf16(
                false, a, false, b, (short)0, acc[nt], false, false);
        }
    }
    // store: acc[nt][r] -> (m = r0 + r + 8*lh, n = c0 + nt*16 + lm)
    #pragma unroll
    for (int nt = 0; nt < 8; ++nt) {
        int n = c0 + nt * 16 + lm;
        #pragma unroll
        for (int r = 0; r < 8; ++r) {
            int m = r0 + r + 8 * lh;
            if (m < nRows) Cout[(size_t)m * HD + n] = acc[nt][r];
        }
    }
}

// ---------------- edge scatter-add: agg[dst] += hn[src] ----------------
__global__ void scatter_add_kernel(const int* __restrict__ src, const int* __restrict__ dst,
                                   const float* __restrict__ hn, float* __restrict__ agg, int nE) {
    int e = blockIdx.x * 8 + (threadIdx.x >> 5);
    if (e >= nE) return;
    int lane = threadIdx.x & 31;
    int s = src[e], d = dst[e];
    const float4* ps = (const float4*)(hn + (size_t)s * HD) + lane * 2;
    float4 v0 = ps[0], v1 = ps[1];
    float* pd = agg + (size_t)d * HD + lane * 8;
    atomicAdd(pd + 0, v0.x); atomicAdd(pd + 1, v0.y);
    atomicAdd(pd + 2, v0.z); atomicAdd(pd + 3, v0.w);
    atomicAdd(pd + 4, v1.x); atomicAdd(pd + 5, v1.y);
    atomicAdd(pd + 6, v1.z); atomicAdd(pd + 7, v1.w);
}

// hr = relu(agg * rsqrt(max(deg_in,1)) + b_conv)   (in place on agg)
__global__ void post_conv_kernel(float* __restrict__ agg, const float* __restrict__ deg_in,
                                 const float* __restrict__ b_conv) {
    int row = blockIdx.x;
    int col = threadIdx.x;
    size_t idx = (size_t)row * HD + col;
    float v = agg[idx] * rsqrtf(fmaxf(deg_in[row], 1.0f)) + b_conv[col];
    agg[idx] = fmaxf(v, 0.0f);
}

// gate[n] = hr[n,:] . w_gate + b_gate ; atomicMax per-graph (ordered-uint key)
__global__ void gate_kernel(const float* __restrict__ hr, const float* __restrict__ wg,
                            const float* __restrict__ bg, const int* __restrict__ graph_id,
                            float* __restrict__ gate, unsigned* __restrict__ gmax, int nN) {
    int n = blockIdx.x * 8 + (threadIdx.x >> 5);
    if (n >= nN) return;
    int lane = threadIdx.x & 31;
    const float* p = hr + (size_t)n * HD + lane * 8;
    const float* q = wg + lane * 8;
    float s = 0.0f;
    #pragma unroll
    for (int i = 0; i < 8; ++i) s += p[i] * q[i];
    #pragma unroll
    for (int off = 16; off > 0; off >>= 1) s += __shfl_xor(s, off, 32);
    if (lane == 0) {
        float g = s + bg[0];
        gate[n] = g;
        int bits = __float_as_int(g);
        unsigned key = (bits >= 0) ? ((unsigned)bits + 0x80000000u) : ~(unsigned)bits;
        atomicMax(&gmax[graph_id[n]], key);
    }
}

__global__ void exp_denom_kernel(const float* __restrict__ gate, const int* __restrict__ graph_id,
                                 const unsigned* __restrict__ gmax, float* __restrict__ expv,
                                 float* __restrict__ denom, int nN) {
    int n = blockIdx.x * blockDim.x + threadIdx.x;
    if (n >= nN) return;
    int g = graph_id[n];
    unsigned key = gmax[g];
    int bits = (key >= 0x80000000u) ? (int)(key - 0x80000000u) : (int)~key;
    float m = __int_as_float(bits);
    float e = __expf(gate[n] - m);
    expv[n] = e;
    atomicAdd(&denom[g], e);
}

// att[n] = expv/denom ; hg[g,:] += att * hr[n,:]
__global__ void att_hg_kernel(const float* __restrict__ hr, const float* __restrict__ expv,
                              const float* __restrict__ denom, const int* __restrict__ graph_id,
                              float* __restrict__ att_out, float* __restrict__ hg, int nN) {
    int n = blockIdx.x * 8 + (threadIdx.x >> 5);
    if (n >= nN) return;
    int lane = threadIdx.x & 31;
    int g = graph_id[n];
    float att = expv[n] / denom[g];
    if (lane == 0) att_out[n] = att;
    const float* p = hr + (size_t)n * HD + lane * 8;
    float* q = hg + (size_t)g * HD + lane * 8;
    #pragma unroll
    for (int i = 0; i < 8; ++i) atomicAdd(q + i, att * p[i]);
}

// hg f32 -> output slot + bf16 copy for head GEMM
__global__ void hg_out_kernel(const float* __restrict__ hg, float* __restrict__ out_hg,
                              __bf16* __restrict__ hgbf) {
    int i = blockIdx.x * blockDim.x + threadIdx.x;   // 0 .. GRAPHS*HD-1
    float v = hg[i];
    out_hg[i] = v;
    hgbf[i] = (__bf16)v;
}

// ---------------- WMMA GEMM2: a2 = hgbf @ W1t^T + b1   (64 x 256) ----------------
// single block, 16 waves: wave w -> M tile (w&3), col range (w>>2)*64 .. +63
__global__ __launch_bounds__(512)
void gemm_head_wmma(const __bf16* __restrict__ A, const __bf16* __restrict__ Bt,
                    const float* __restrict__ bias, float* __restrict__ Cout) {
    const int tid  = threadIdx.x;
    const int lane = tid & 31;
    const int w    = tid >> 5;
    const int r0   = (w & 3) * 16;
    const int c0   = (w >> 2) * 64;
    const int lh   = lane >> 4;
    const int lm   = lane & 15;

    v8f acc[4] = {};

    const __bf16* aRow = A + (size_t)(r0 + lm) * HD;
    #pragma unroll
    for (int k0 = 0; k0 < HD; k0 += 32) {
        AFrag af;
        af.h.lo = *(const bf16x8*)(aRow + k0 + lh * 16);
        af.h.hi = *(const bf16x8*)(aRow + k0 + lh * 16 + 8);
        v16bf a = af.v;
        #pragma unroll
        for (int nt = 0; nt < 4; ++nt) {
            const __bf16* bp = Bt + (size_t)(c0 + nt * 16 + lm) * HD + k0 + lh * 16;
            v16bf b = *(const v16bf*)bp;
            acc[nt] = __builtin_amdgcn_wmma_f32_16x16x32_bf16(
                false, a, false, b, (short)0, acc[nt], false, false);
        }
    }
    #pragma unroll
    for (int nt = 0; nt < 4; ++nt) {
        int n = c0 + nt * 16 + lm;
        float bv = bias[n];
        #pragma unroll
        for (int r = 0; r < 8; ++r) {
            int m = r0 + r + 8 * lh;
            Cout[(size_t)m * HD + n] = acc[nt][r] + bv;
        }
    }
}

// a3 = sigmoid(a2 @ W2 + b2) -> out[0..127]
__global__ void head2_kernel(const float* __restrict__ a2, const float* __restrict__ W2,
                             const float* __restrict__ b2, float* __restrict__ out) {
    int t = threadIdx.x;                 // 0..127
    int g = t >> 1, c = t & 1;
    float s = b2[c];
    const float* p = a2 + (size_t)g * HD;
    #pragma unroll 8
    for (int k = 0; k < HD; ++k) s += p[k] * W2[k * NCLS + c];
    out[t] = 1.0f / (1.0f + __expf(-s));
}

// ---------------- host orchestration ----------------
extern "C" void kernel_launch(void* const* d_in, const int* in_sizes, int n_in,
                              void* d_out, int out_size, void* d_ws, size_t ws_size,
                              hipStream_t stream) {
    const float* h        = (const float*)d_in[0];
    const int*   src      = (const int*)  d_in[1];
    const int*   dst      = (const int*)  d_in[2];
    const int*   graph_id = (const int*)  d_in[3];
    const float* W_conv   = (const float*)d_in[4];
    const float* b_conv   = (const float*)d_in[5];
    const float* w_gate   = (const float*)d_in[6];
    const float* b_gate   = (const float*)d_in[7];
    const float* W1       = (const float*)d_in[8];
    const float* b1       = (const float*)d_in[9];
    const float* W2       = (const float*)d_in[10];
    const float* b2       = (const float*)d_in[11];
    float* out = (float*)d_out;
    // output layout: [sigmoid(a3): 128][att: 20000][hg: 16384]
    float* out_sig = out;
    float* out_att = out + GRAPHS * NCLS;
    float* out_hg  = out + GRAPHS * NCLS + NODES;

    // carve workspace (256B-aligned chunks)
    char* p = (char*)d_ws;
    auto carve = [&](size_t bytes) -> char* {
        char* r = p;
        p += (bytes + 255) & ~(size_t)255;
        return r;
    };
    float*    deg_out = (float*)   carve(sizeof(float) * NODES);
    float*    deg_in  = (float*)   carve(sizeof(float) * NODES);
    __bf16*   hbf     = (__bf16*)  carve(sizeof(__bf16) * (size_t)NPAD * HD);
    __bf16*   Wct     = (__bf16*)  carve(sizeof(__bf16) * HD * HD);
    __bf16*   W1t     = (__bf16*)  carve(sizeof(__bf16) * HD * HD);
    float*    hn      = (float*)   carve(sizeof(float) * (size_t)NODES * HD);
    float*    agg     = (float*)   carve(sizeof(float) * (size_t)NODES * HD); // becomes hr
    float*    gate    = (float*)   carve(sizeof(float) * NODES);
    float*    expv    = (float*)   carve(sizeof(float) * NODES);
    // contiguous small region zeroed together: gmax | denom | hg
    float*    smallz  = (float*)   carve(sizeof(float) * (GRAPHS + GRAPHS + GRAPHS * HD));
    unsigned* gmax    = (unsigned*)smallz;
    float*    denom   = smallz + GRAPHS;
    float*    hg      = smallz + 2 * GRAPHS;
    __bf16*   hgbf    = (__bf16*)  carve(sizeof(__bf16) * GRAPHS * HD);
    float*    a2      = (float*)   carve(sizeof(float) * GRAPHS * HD);

    // 1) zero accumulators (every call: harness does not re-poison)
    zero_f32<<<(2 * NODES + 255) / 256, 256, 0, stream>>>(deg_out, 2 * NODES); // deg_out+deg_in contiguous? no -> do both
    zero_f32<<<(NODES + 255) / 256, 256, 0, stream>>>(deg_in, NODES);
    zero_f32<<<(NODES * HD + 255) / 256, 256, 0, stream>>>(agg, NODES * HD);
    zero_f32<<<((2 * GRAPHS + GRAPHS * HD) + 255) / 256, 256, 0, stream>>>(smallz, 2 * GRAPHS + GRAPHS * HD);

    // 2) degrees
    degree_kernel<<<(EDGES + 255) / 256, 256, 0, stream>>>(src, dst, deg_out, deg_in, EDGES);

    // 3) precision conversion / layout transform
    convert_h_kernel<<<NPAD, HD, 0, stream>>>(h, deg_out, hbf);
    convert_w_kernel<<<(2 * HD * HD + 255) / 256, 256, 0, stream>>>(W_conv, W1, Wct, W1t);

    // 4) GCN weight GEMM (WMMA bf16 -> f32)
    gemm_conv_wmma<<<GEMM1_BLOCKS, 512, 0, stream>>>(hbf, Wct, hn, NODES);

    // 5) message passing: scatter-add along edges
    scatter_add_kernel<<<(EDGES + 7) / 8, 256, 0, stream>>>(src, dst, hn, agg, EDGES);

    // 6) normalize + bias + relu -> hr (in place)
    post_conv_kernel<<<NODES, HD, 0, stream>>>(agg, deg_in, b_conv);

    // 7) attention gate + segment softmax
    gate_kernel<<<(NODES + 7) / 8, 256, 0, stream>>>(agg, w_gate, b_gate, graph_id, gate, gmax, NODES);
    exp_denom_kernel<<<(NODES + 255) / 256, 256, 0, stream>>>(gate, graph_id, gmax, expv, denom, NODES);
    att_hg_kernel<<<(NODES + 7) / 8, 256, 0, stream>>>(agg, expv, denom, graph_id, out_att, hg, NODES);

    // 8) emit hg + convert for head GEMM
    hg_out_kernel<<<(GRAPHS * HD + 255) / 256, 256, 0, stream>>>(hg, out_hg, hgbf);

    // 9) classifier heads
    gemm_head_wmma<<<1, 512, 0, stream>>>(hgbf, W1t, b1, a2);
    head2_kernel<<<1, 128, 0, stream>>>(a2, W2, b2, out_sig);
}
